// Net_64364379898496
// MI455X (gfx1250) — compile-verified, matrix operand order checked by reading, not run on hardware
//
#include <hip/hip_runtime.h>
#include <stdint.h>

// ---------------------------------------------------------------------------
// Problem constants (from reference)
// ---------------------------------------------------------------------------
#define T_STEPS 50
#define B_SZ    1024
#define N_IN    784
#define N_HID   1024
#define N_OUT   10
#define K_PAD   800                     // 784 padded to multiple of 32
#define M_TOT   (T_STEPS * B_SZ)        // 51200

typedef __attribute__((ext_vector_type(16))) __bf16 v16bf;
typedef __attribute__((ext_vector_type(8)))  float  v8f;

// ---------------------------------------------------------------------------
// bf16 helpers (RNE)
// ---------------------------------------------------------------------------
__device__ __forceinline__ unsigned short f32_to_bf16(float f) {
  unsigned int u = __float_as_uint(f);
  unsigned int lsb = (u >> 16) & 1u;
  u += 0x7fffu + lsb;
  return (unsigned short)(u >> 16);
}
__device__ __forceinline__ float bf16_to_f32(unsigned short h) {
  return __uint_as_float(((unsigned int)h) << 16);
}

// CDNA5 async memory->LDS DMA (ASYNCcnt-tracked), 16B per lane.
// LDS dest = low 32 bits of generic pointer (aperture lives in high dword).
__device__ __forceinline__ void async_ld16(unsigned lds_byte_off, const void* gaddr) {
  asm volatile("global_load_async_to_lds_b128 %0, %1, off"
               :: "v"(lds_byte_off), "v"(gaddr)
               : "memory");
}

// ---------------------------------------------------------------------------
// Split fp32 -> bf16 hi/lo, pad K 784 -> 800 with zeros
// ---------------------------------------------------------------------------
__global__ __launch_bounds__(256) void cvtX_kernel(
    const float* __restrict__ X, unsigned short* __restrict__ hi,
    unsigned short* __restrict__ lo) {
  const int idx = blockIdx.x * 256 + threadIdx.x;   // M_TOT*K_PAD threads
  const int m = idx / K_PAD;
  const int k = idx - m * K_PAD;
  float v = (k < N_IN) ? X[(size_t)m * N_IN + k] : 0.0f;
  unsigned short h = f32_to_bf16(v);
  hi[idx] = h;
  lo[idx] = f32_to_bf16(v - bf16_to_f32(h));
}

__global__ __launch_bounds__(256) void cvtW_kernel(
    const float* __restrict__ W, unsigned short* __restrict__ hi,
    unsigned short* __restrict__ lo) {
  const int idx = blockIdx.x * 256 + threadIdx.x;   // N_HID*K_PAD threads
  const int n = idx / K_PAD;
  const int k = idx - n * K_PAD;
  float v = (k < N_IN) ? W[(size_t)n * N_IN + k] : 0.0f;
  unsigned short h = f32_to_bf16(v);
  hi[idx] = h;
  lo[idx] = f32_to_bf16(v - bf16_to_f32(h));
}

// ---------------------------------------------------------------------------
// GEMM: CUR1[M_TOT x N_HID] = X * W1^T  (bf16 hi/lo split, f32 accumulate)
// Block tile 128x128, 8 waves (4x2), wave tile 32x64 (2x4 WMMA tiles).
// Double-buffered async global->LDS staging (ASYNCcnt).
// ---------------------------------------------------------------------------
#define BM 128
#define BN 128
#define BK 32
#define NK (K_PAD / BK)   // 25
#define LDA 56   // halves per LDS row: 112B (16B aligned, 7r mod 16 permutes banks)

union Frag {
  v16bf bf;
  uint4 q[2];
};

__global__ __launch_bounds__(256) void gemm1_kernel(
    const unsigned short* __restrict__ Ahi, const unsigned short* __restrict__ Alo,
    const unsigned short* __restrict__ Bhi, const unsigned short* __restrict__ Blo,
    float* __restrict__ C) {
  __shared__ unsigned short sAh[2][BM * LDA];
  __shared__ unsigned short sAl[2][BM * LDA];
  __shared__ unsigned short sBh[2][BN * LDA];
  __shared__ unsigned short sBl[2][BN * LDA];

  const int tid  = threadIdx.x;
  const int lane = tid & 31;
  const int wave = tid >> 5;
  const int wm = wave & 3;          // 4 waves along M (x32)
  const int wn = wave >> 2;         // 2 waves along N (x64)
  const int m0 = blockIdx.x * BM;
  const int n0 = blockIdx.y * BN;

  v8f acc[2][4] = {};

  // WMMA 16-bit A/B fragment lane addressing (ISA 7.12.2):
  // lane<16: row=lane,  K slices [0..7] and [16..23]
  // lane>=16: row=lane-16, K slices [8..15] and [24..31]
  const int frow = lane & 15;
  const int fk   = (lane >> 4) << 3;   // 0 or 8

  // staging map: thread t covers row sr, halves [sc..sc+15] (two 16B chunks)
  const int sr = tid >> 1;             // 0..127
  const int sc = (tid & 1) << 4;       // 0 or 16 halves

  auto stage = [&](int buf, int k0) {
    const unsigned short* ga_h = Ahi + (size_t)(m0 + sr) * K_PAD + k0 + sc;
    const unsigned short* ga_l = Alo + (size_t)(m0 + sr) * K_PAD + k0 + sc;
    const unsigned short* gb_h = Bhi + (size_t)(n0 + sr) * K_PAD + k0 + sc;
    const unsigned short* gb_l = Blo + (size_t)(n0 + sr) * K_PAD + k0 + sc;
    const unsigned la_h = (unsigned)(uintptr_t)(sAh[buf] + sr * LDA + sc);
    const unsigned la_l = (unsigned)(uintptr_t)(sAl[buf] + sr * LDA + sc);
    const unsigned lb_h = (unsigned)(uintptr_t)(sBh[buf] + sr * LDA + sc);
    const unsigned lb_l = (unsigned)(uintptr_t)(sBl[buf] + sr * LDA + sc);
    async_ld16(la_h,      ga_h);
    async_ld16(la_h + 16, ga_h + 8);
    async_ld16(la_l,      ga_l);
    async_ld16(la_l + 16, ga_l + 8);
    async_ld16(lb_h,      gb_h);
    async_ld16(lb_h + 16, gb_h + 8);
    async_ld16(lb_l,      gb_l);
    async_ld16(lb_l + 16, gb_l + 8);
  };

  stage(0, 0);  // prologue: 8 async loads outstanding

  for (int it = 0; it < NK; ++it) {
    const int cur = it & 1;
    if (it + 1 < NK) {
      stage(cur ^ 1, (it + 1) * BK);
      // 8 newly issued may remain in flight; the 8 for buf[cur] are done
      asm volatile("s_wait_asynccnt 0x8" ::: "memory");
    } else {
      asm volatile("s_wait_asynccnt 0x0" ::: "memory");
    }
    __syncthreads();   // buf[cur] visible to all waves

    const unsigned short* pAh = sAh[cur];
    const unsigned short* pAl = sAl[cur];
    const unsigned short* pBh = sBh[cur];
    const unsigned short* pBl = sBl[cur];

    Frag ah[2], al[2], bh[4], bl[4];
#pragma unroll
    for (int mt = 0; mt < 2; ++mt) {
      const int base = (wm * 32 + mt * 16 + frow) * LDA + fk;
      ah[mt].q[0] = *(const uint4*)(pAh + base);
      ah[mt].q[1] = *(const uint4*)(pAh + base + 16);
      al[mt].q[0] = *(const uint4*)(pAl + base);
      al[mt].q[1] = *(const uint4*)(pAl + base + 16);
    }
#pragma unroll
    for (int nt = 0; nt < 4; ++nt) {
      const int base = (wn * 64 + nt * 16 + frow) * LDA + fk;
      bh[nt].q[0] = *(const uint4*)(pBh + base);
      bh[nt].q[1] = *(const uint4*)(pBh + base + 16);
      bl[nt].q[0] = *(const uint4*)(pBl + base);
      bl[nt].q[1] = *(const uint4*)(pBl + base + 16);
    }

    // hi/lo split products: hi*hi + hi*lo + lo*hi (f32 accumulate)
#pragma unroll
    for (int mt = 0; mt < 2; ++mt)
#pragma unroll
      for (int nt = 0; nt < 4; ++nt) {
        acc[mt][nt] = __builtin_amdgcn_wmma_f32_16x16x32_bf16(
            false, ah[mt].bf, false, bh[nt].bf, (short)0, acc[mt][nt], false, false);
        acc[mt][nt] = __builtin_amdgcn_wmma_f32_16x16x32_bf16(
            false, ah[mt].bf, false, bl[nt].bf, (short)0, acc[mt][nt], false, false);
        acc[mt][nt] = __builtin_amdgcn_wmma_f32_16x16x32_bf16(
            false, al[mt].bf, false, bh[nt].bf, (short)0, acc[mt][nt], false, false);
      }
    __syncthreads();   // all waves done reading buf[cur] before it is restaged
  }

  // store C per documented 16x16 f32 D layout
  const int rbase = (lane < 16) ? 0 : 8;
  const int ccol  = lane & 15;
#pragma unroll
  for (int mt = 0; mt < 2; ++mt)
#pragma unroll
    for (int nt = 0; nt < 4; ++nt)
#pragma unroll
      for (int r = 0; r < 8; ++r) {
        const int row = m0 + wm * 32 + mt * 16 + rbase + r;
        const int col = n0 + wn * 64 + nt * 16 + ccol;
        C[(size_t)row * N_HID + col] = acc[mt][nt][r];
      }
}

// ---------------------------------------------------------------------------
// LIF layer 1: mem1 recursion over t; bit-pack spikes with wave32 ballot
// ---------------------------------------------------------------------------
__global__ __launch_bounds__(256) void lif1_kernel(
    const float* __restrict__ cur1, const float* __restrict__ beta1,
    unsigned int* __restrict__ spk) {
  const int gid  = blockIdx.x * 256 + threadIdx.x;  // B_SZ*N_HID threads
  const int b    = gid >> 10;
  const int h    = gid & 1023;
  const int lane = threadIdx.x & 31;
  const float bb = fminf(fmaxf(*beta1, 0.0f), 1.0f);
  float mem = 0.0f;
  for (int t = 0; t < T_STEPS; ++t) {
    const float c = cur1[((size_t)(t * B_SZ + b)) * N_HID + h];
    const bool reset = mem > 1.0f;               // reset from previous mem
    mem = reset ? 0.0f : fmaf(bb, mem, c);
    const unsigned int mask = __builtin_amdgcn_ballot_w32(mem > 1.0f);
    if (lane == 0) spk[(t * B_SZ + b) * 32 + (h >> 5)] = mask;
  }
}

// ---------------------------------------------------------------------------
// LIF layer 2: cur2 from bit-packed spikes (exact), mem2 recursion, outputs
// ---------------------------------------------------------------------------
__global__ __launch_bounds__(256) void lif2_kernel(
    const unsigned int* __restrict__ spk, const float* __restrict__ W2,
    const float* __restrict__ beta2, float* __restrict__ out) {
  const int gid = blockIdx.x * 256 + threadIdx.x;  // B_SZ*N_OUT threads (10240)
  const int b = gid / N_OUT;
  const int o = gid - b * N_OUT;
  const float bb = fminf(fmaxf(*beta2, 0.0f), 1.0f);
  const float* __restrict__ w = W2 + o * N_HID;
  float mem = 0.0f;
  for (int t = 0; t < T_STEPS; ++t) {
    const unsigned int* __restrict__ sp = spk + (t * B_SZ + b) * 32;
    float cur = 0.0f;
    for (int wd = 0; wd < 32; ++wd) {
      const unsigned int bits = sp[wd];
      const float* __restrict__ wp = w + wd * 32;
#pragma unroll
      for (int j = 0; j < 32; ++j)
        cur += ((bits >> j) & 1u) ? wp[j] : 0.0f;
    }
    const bool reset = mem > 1.0f;
    mem = reset ? 0.0f : fmaf(bb, mem, cur);
    const float s2 = (mem > 1.0f) ? 1.0f : 0.0f;
    const int idx = (t * B_SZ + b) * N_OUT + o;
    out[idx] = s2;
    out[T_STEPS * B_SZ * N_OUT + idx] = mem;
  }
}

// ---------------------------------------------------------------------------
// Host-side launcher
// ---------------------------------------------------------------------------
extern "C" void kernel_launch(void* const* d_in, const int* in_sizes, int n_in,
                              void* d_out, int out_size, void* d_ws, size_t ws_size,
                              hipStream_t stream) {
  (void)in_sizes; (void)n_in; (void)out_size; (void)ws_size;
  const float* X  = (const float*)d_in[0];
  const float* W1 = (const float*)d_in[1];
  const float* W2 = (const float*)d_in[2];
  const float* b1 = (const float*)d_in[3];
  const float* b2 = (const float*)d_in[4];
  float* out = (float*)d_out;

  // workspace carve-out (256B aligned blocks); total ~384 MB
  char* ws = (char*)d_ws;
  size_t off = 0;
  auto carve = [&](size_t bytes) -> void* {
    void* p = ws + off;
    off += (bytes + 255) & ~(size_t)255;
    return p;
  };
  unsigned short* Xhi  = (unsigned short*)carve((size_t)M_TOT * K_PAD * 2);
  unsigned short* Xlo  = (unsigned short*)carve((size_t)M_TOT * K_PAD * 2);
  unsigned short* W1hi = (unsigned short*)carve((size_t)N_HID * K_PAD * 2);
  unsigned short* W1lo = (unsigned short*)carve((size_t)N_HID * K_PAD * 2);
  float*          CUR1 = (float*)carve((size_t)M_TOT * N_HID * 4);
  unsigned int*   SPK  = (unsigned int*)carve((size_t)T_STEPS * B_SZ * 32 * 4);

  cvtX_kernel<<<(M_TOT * K_PAD) / 256, 256, 0, stream>>>(X, Xhi, Xlo);
  cvtW_kernel<<<(N_HID * K_PAD) / 256, 256, 0, stream>>>(W1, W1hi, W1lo);
  gemm1_kernel<<<dim3(M_TOT / BM, N_HID / BN), 256, 0, stream>>>(
      Xhi, Xlo, W1hi, W1lo, CUR1);
  lif1_kernel<<<(B_SZ * N_HID) / 256, 256, 0, stream>>>(CUR1, b1, SPK);
  lif2_kernel<<<(B_SZ * N_OUT) / 256, 256, 0, stream>>>(SPK, W2, b2, out);
}
